// Encoder_43035572306780
// MI455X (gfx1250) — matrix operand order repaired and checked by reading, not compiled, not run on hardware
//
#include <hip/hip_runtime.h>
#include <hip/hip_bf16.h>

// ---------------------------------------------------------------------------
// Problem constants (from reference)
// ---------------------------------------------------------------------------
static constexpr int Bb = 128;   // batch
static constexpr int Tt = 512;   // timesteps
static constexpr int Dd = 128;   // input features
static constexpr int Hh = 512;   // hidden
static constexpr int Ll = 128;   // latent
static constexpr int G  = 4 * Hh; // 2048 gates

typedef __attribute__((ext_vector_type(16))) __bf16 v16bf;
typedef __attribute__((ext_vector_type(8)))  __bf16 v8bf;
typedef __attribute__((ext_vector_type(8)))  float  v8f;

// ---------------------------------------------------------------------------
// Workspace layout (bytes). All segments 256B-aligned by construction.
// ---------------------------------------------------------------------------
static constexpr size_t SZ_XBF  = (size_t)Bb * Tt * Dd * 2;      // bf16 inputs
static constexpr size_t OFF_XBF  = 0;
static constexpr size_t OFF_W1XT = OFF_XBF  + SZ_XBF;            // [G][D] bf16
static constexpr size_t OFF_W1HT = OFF_W1XT + (size_t)G * Dd * 2; // [G][H]
static constexpr size_t OFF_W2XT = OFF_W1HT + (size_t)G * Hh * 2;
static constexpr size_t OFF_W2HT = OFF_W2XT + (size_t)G * Hh * 2;
static constexpr size_t OFF_W3XT = OFF_W2HT + (size_t)G * Hh * 2;
static constexpr size_t OFF_W3HT = OFF_W3XT + (size_t)G * Hh * 2;
static constexpr size_t OFF_H0   = OFF_W3HT + (size_t)G * Hh * 2; // h1,h2,h3 parity0 (bf16)
static constexpr size_t OFF_C    = OFF_H0 + 3 * (size_t)Bb * Hh * 2; // c1,c2,c3 (f32)
static constexpr size_t OFF_H1   = OFF_C  + 3 * (size_t)Bb * Hh * 4; // h parity1
static constexpr size_t OFF_LOSS = OFF_H1 + 3 * (size_t)Bb * Hh * 2; // B*L f32
// zero region: OFF_H0 .. OFF_H1  (parity-0 h buffers + all c buffers)
static constexpr int ZERO_WORDS = (int)((OFF_H1 - OFF_H0) / 4);

// ---------------------------------------------------------------------------
// Small helper kernels
// ---------------------------------------------------------------------------
__global__ void zero_words_kernel(unsigned int* __restrict__ p, int n) {
  int i = blockIdx.x * blockDim.x + threadIdx.x;
  if (i < n) p[i] = 0u;
}

__global__ void f32_to_bf16_kernel(const float* __restrict__ src,
                                   __bf16* __restrict__ dst, int n) {
  int i = blockIdx.x * blockDim.x + threadIdx.x;
  if (i < n) dst[i] = (__bf16)src[i];
}

// W: [K][2048] f32 row-major  ->  WT: [2048][K] bf16 row-major (K contiguous)
__global__ void transpose_to_bf16_kernel(const float* __restrict__ W,
                                         __bf16* __restrict__ WT, int K) {
  int i = blockIdx.x * blockDim.x + threadIdx.x;
  int total = K * G;
  if (i >= total) return;
  int nrow = i / K;
  int k    = i % K;
  WT[i] = (__bf16)W[(size_t)k * G + nrow];
}

// ---------------------------------------------------------------------------
// Fused LSTM step:  Z = X @ Wx + Hprev @ Wh + b ; gate math ; update c,h
//   X     : [128][ldx] bf16 (rows = batch)
//   WxT   : [2048][Kx] bf16 (pre-transposed, K contiguous)
//   Hprev : [128][512] bf16
//   WhT   : [2048][512] bf16
//   C     : [128][512] f32 (in-place)
//   Hnew  : [128][512] bf16
// Grid (32, 2), block 128 (4 waves). Wave owns 16 rows x 16 h-cols x 4 gates.
// ---------------------------------------------------------------------------
__device__ __forceinline__ float sigmoid_f(float x) {
  return 1.0f / (1.0f + __expf(-x));
}
__device__ __forceinline__ float tanh_f(float x) {
  x = fminf(fmaxf(x, -15.0f), 15.0f);
  float e = __expf(2.0f * x);
  return (e - 1.0f) / (e + 1.0f);
}

__launch_bounds__(128, 1)
__global__ void lstm_step_kernel(const __bf16* __restrict__ X, int ldx, int Kx,
                                 const __bf16* __restrict__ WxT,
                                 const __bf16* __restrict__ Hprev,
                                 const __bf16* __restrict__ WhT,
                                 const float* __restrict__ bias,
                                 float* __restrict__ C,
                                 __bf16* __restrict__ Hnew) {
  const int lane = threadIdx.x & 31;
  const int wave = threadIdx.x >> 5;
  const int m0 = blockIdx.y * 64 + wave * 16;   // row tile (batch)
  const int n0 = blockIdx.x * 16;               // h-column tile
  const int r15 = lane & 15;
  const int kAoff = (lane & 16);                // A-matrix intra-lane K half
  const int kBoff = (lane & 16);                // B-matrix intra-lane K half

  v8f acc[4];
#pragma unroll
  for (int q = 0; q < 4; ++q)
    acc[q] = (v8f){0.f, 0.f, 0.f, 0.f, 0.f, 0.f, 0.f, 0.f};

  // Per-lane base pointers (A: row = m0+r15 ; B: row = gate*512 + n0 + r15)
  const __bf16* Ax = X + (size_t)(m0 + r15) * ldx;
  const __bf16* Ah = Hprev + (size_t)(m0 + r15) * Hh;
  const __bf16* Bx[4];
  const __bf16* Bh[4];
#pragma unroll
  for (int q = 0; q < 4; ++q) {
    Bx[q] = WxT + (size_t)(q * Hh + n0 + r15) * Kx + kBoff;
    Bh[q] = WhT + (size_t)(q * Hh + n0 + r15) * Hh + kBoff;
  }

  // ---- pass 1: X @ Wx ----
  for (int kb = 0; kb < Kx; kb += 32) {
    v8bf alo = *(const v8bf*)(Ax + kb + kAoff);
    v8bf ahi = *(const v8bf*)(Ax + kb + 8 + kAoff);
    v16bf a = __builtin_shufflevector(alo, ahi, 0, 1, 2, 3, 4, 5, 6, 7,
                                      8, 9, 10, 11, 12, 13, 14, 15);
#pragma unroll
    for (int q = 0; q < 4; ++q) {
      v16bf b = *(const v16bf*)(Bx[q] + kb);
      acc[q] = __builtin_amdgcn_wmma_f32_16x16x32_bf16(
          false, a, false, b, (short)0, acc[q], false, false);
    }
  }

  // ---- pass 2: Hprev @ Wh ----
  for (int kb = 0; kb < Hh; kb += 32) {
    v8bf alo = *(const v8bf*)(Ah + kb + kAoff);
    v8bf ahi = *(const v8bf*)(Ah + kb + 8 + kAoff);
    v16bf a = __builtin_shufflevector(alo, ahi, 0, 1, 2, 3, 4, 5, 6, 7,
                                      8, 9, 10, 11, 12, 13, 14, 15);
#pragma unroll
    for (int q = 0; q < 4; ++q) {
      v16bf b = *(const v16bf*)(Bh[q] + kb);
      acc[q] = __builtin_amdgcn_wmma_f32_16x16x32_bf16(
          false, a, false, b, (short)0, acc[q], false, false);
    }
  }

  // ---- bias (per output column) ----
  const int ncol = n0 + r15;
  float bv[4];
#pragma unroll
  for (int q = 0; q < 4; ++q) bv[q] = bias[q * Hh + ncol];

  // ---- fused LSTM pointwise update ----
  // C/D layout: VGPR r -> M = r (+8 for lanes 16..31), N = lane&15.
  const int mbase = m0 + ((lane & 16) ? 8 : 0);
#pragma unroll
  for (int r = 0; r < 8; ++r) {
    const int m = mbase + r;
    const size_t idx = (size_t)m * Hh + ncol;
    float zi = acc[0][r] + bv[0];
    float zf = acc[1][r] + bv[1];
    float zg = acc[2][r] + bv[2];
    float zo = acc[3][r] + bv[3];
    float c_old = C[idx];
    float c_new = sigmoid_f(zf) * c_old + sigmoid_f(zi) * tanh_f(zg);
    float h_new = sigmoid_f(zo) * tanh_f(c_new);
    C[idx] = c_new;
    Hnew[idx] = (__bf16)h_new;
  }
}

// ---------------------------------------------------------------------------
// Heads: mean = c3 @ w_mean + b_mean ; sigma likewise ;
// retval = mean + exp(sigma/2) * noise ; per-element loss term to lossbuf.
// ---------------------------------------------------------------------------
__global__ void head_kernel(const float* __restrict__ c3,
                            const float* __restrict__ wm,
                            const float* __restrict__ bm,
                            const float* __restrict__ wsg,
                            const float* __restrict__ bsg,
                            const float* __restrict__ noise,
                            float* __restrict__ out,
                            float* __restrict__ lossbuf) {
  int idx = blockIdx.x * blockDim.x + threadIdx.x;   // 0..B*L-1
  if (idx >= Bb * Ll) return;
  int b = idx >> 7;       // / L
  int l = idx & (Ll - 1); // % L
  const float* crow = c3 + (size_t)b * Hh;
  float m = bm[l], s = bsg[l];
  for (int k = 0; k < Hh; ++k) {
    float cv = crow[k];
    m += cv * wm[(size_t)k * Ll + l];
    s += cv * wsg[(size_t)k * Ll + l];
  }
  const float NOISE_MEAN = 0.0f, NOISE_STD = 1.0f;
  float eps = NOISE_MEAN + NOISE_STD * noise[idx];
  out[idx] = m + __expf(0.5f * s) * eps;
  lossbuf[idx] = -0.5f * (1.0f + s - m * m - __expf(s));
}

__global__ void loss_kernel(const float* __restrict__ lossbuf,
                            float* __restrict__ out) {
  __shared__ float red[256];
  float acc = 0.0f;
  for (int i = threadIdx.x; i < Bb * Ll; i += 256) acc += lossbuf[i];
  red[threadIdx.x] = acc;
  __syncthreads();
  for (int s = 128; s > 0; s >>= 1) {
    if ((int)threadIdx.x < s) red[threadIdx.x] += red[threadIdx.x + s];
    __syncthreads();
  }
  if (threadIdx.x == 0) out[Bb * Ll] = red[0] / (float)(Bb * Ll);
}

// ---------------------------------------------------------------------------
// Host driver
// ---------------------------------------------------------------------------
extern "C" void kernel_launch(void* const* d_in, const int* in_sizes, int n_in,
                              void* d_out, int out_size, void* d_ws, size_t ws_size,
                              hipStream_t stream) {
  (void)in_sizes; (void)n_in; (void)out_size; (void)ws_size;

  const float* inputs = (const float*)d_in[0];
  const float* noise  = (const float*)d_in[1];
  const float* W1x = (const float*)d_in[2];
  const float* W1h = (const float*)d_in[3];
  const float* b1  = (const float*)d_in[4];
  const float* W2x = (const float*)d_in[5];
  const float* W2h = (const float*)d_in[6];
  const float* b2  = (const float*)d_in[7];
  const float* W3x = (const float*)d_in[8];
  const float* W3h = (const float*)d_in[9];
  const float* b3  = (const float*)d_in[10];
  const float* wm  = (const float*)d_in[11];
  const float* bm  = (const float*)d_in[12];
  const float* wsg = (const float*)d_in[13];
  const float* bsg = (const float*)d_in[14];

  char* ws = (char*)d_ws;
  __bf16* xbf  = (__bf16*)(ws + OFF_XBF);
  __bf16* w1xt = (__bf16*)(ws + OFF_W1XT);
  __bf16* w1ht = (__bf16*)(ws + OFF_W1HT);
  __bf16* w2xt = (__bf16*)(ws + OFF_W2XT);
  __bf16* w2ht = (__bf16*)(ws + OFF_W2HT);
  __bf16* w3xt = (__bf16*)(ws + OFF_W3XT);
  __bf16* w3ht = (__bf16*)(ws + OFF_W3HT);
  float*  cbuf = (float*)(ws + OFF_C);
  float*  lossb = (float*)(ws + OFF_LOSS);
  float* out = (float*)d_out;

  auto hbuf = [&](int layer, int parity) -> __bf16* {
    return (__bf16*)(ws + (parity ? OFF_H1 : OFF_H0) +
                     (size_t)layer * Bb * Hh * 2);
  };
  float* c1 = cbuf;
  float* c2 = cbuf + (size_t)Bb * Hh;
  float* c3 = cbuf + 2 * (size_t)Bb * Hh;

  // --- one-time-per-launch prep: bf16 conversion + weight transposes + zeroing ---
  {
    int n = Bb * Tt * Dd;
    f32_to_bf16_kernel<<<(n + 255) / 256, 256, 0, stream>>>(inputs, xbf, n);
  }
  transpose_to_bf16_kernel<<<(Dd * G + 255) / 256, 256, 0, stream>>>(W1x, w1xt, Dd);
  transpose_to_bf16_kernel<<<(Hh * G + 255) / 256, 256, 0, stream>>>(W1h, w1ht, Hh);
  transpose_to_bf16_kernel<<<(Hh * G + 255) / 256, 256, 0, stream>>>(W2x, w2xt, Hh);
  transpose_to_bf16_kernel<<<(Hh * G + 255) / 256, 256, 0, stream>>>(W2h, w2ht, Hh);
  transpose_to_bf16_kernel<<<(Hh * G + 255) / 256, 256, 0, stream>>>(W3x, w3xt, Hh);
  transpose_to_bf16_kernel<<<(Hh * G + 255) / 256, 256, 0, stream>>>(W3h, w3ht, Hh);
  zero_words_kernel<<<(ZERO_WORDS + 255) / 256, 256, 0, stream>>>(
      (unsigned int*)(ws + OFF_H0), ZERO_WORDS);

  // --- recurrence: 512 steps x 3 layers, each a fused WMMA cell kernel ---
  const dim3 sgrid(32, 2);
  for (int t = 0; t < Tt; ++t) {
    const int pr = t & 1;
    const int pw = pr ^ 1;
    __bf16* h1w = hbuf(0, pw);
    __bf16* h2w = hbuf(1, pw);
    __bf16* h3w = hbuf(2, pw);
    // layer 1: x_t from converted inputs, row stride T*D
    lstm_step_kernel<<<sgrid, 128, 0, stream>>>(
        xbf + (size_t)t * Dd, Tt * Dd, Dd, w1xt, hbuf(0, pr), w1ht, b1, c1, h1w);
    // layer 2: x = h1(t)
    lstm_step_kernel<<<sgrid, 128, 0, stream>>>(
        h1w, Hh, Hh, w2xt, hbuf(1, pr), w2ht, b2, c2, h2w);
    // layer 3: x = h2(t)
    lstm_step_kernel<<<sgrid, 128, 0, stream>>>(
        h2w, Hh, Hh, w3xt, hbuf(2, pr), w3ht, b3, c3, h3w);
  }

  // --- heads + loss on final cell state c3 (fp32) ---
  head_kernel<<<(Bb * Ll + 255) / 256, 256, 0, stream>>>(
      c3, wm, bm, wsg, bsg, noise, out, lossb);
  loss_kernel<<<1, 256, 0, stream>>>(lossb, out);
}
